// Kernel_64793876627487
// MI455X (gfx1250) — matrix operand order repaired and checked by reading, not compiled, not run on hardware
//
#include <hip/hip_runtime.h>
#include <math.h>

// ---------------------------------------------------------------------------
// Problem constants (from the reference)
// ---------------------------------------------------------------------------
#define BATCH   512
#define HIDDEN  64
#define NPATH   70656          // total coefficient columns
#define OUTDIM  480            // 128*1 + 64*3 + 32*5
#define FOURPI  12.566370614359172f

typedef _Float16 half8  __attribute__((ext_vector_type(8)));
typedef _Float16 half16 __attribute__((ext_vector_type(16)));
typedef float    float8 __attribute__((ext_vector_type(8)));

// ---------------------------------------------------------------------------
// Kernel 1: per-batch prep: h = relu(|r|*w1+b1) (as f16), Y = real SH l=0..4
// grid = BATCH blocks of 64 threads (thread k -> hidden unit k)
// ---------------------------------------------------------------------------
__global__ void prep_kernel(const float* __restrict__ r,
                            const float* __restrict__ w1,
                            const float* __restrict__ b1,
                            _Float16* __restrict__ hbuf,
                            float* __restrict__ Ybuf) {
    const int z = blockIdx.x;
    const int k = threadIdx.x;          // 0..63
    const float rx = r[z * 3 + 0];
    const float ry = r[z * 3 + 1];
    const float rz = r[z * 3 + 2];
    const float nrm = sqrtf(rx * rx + ry * ry + rz * rz);

    const float hv = fmaxf(nrm * w1[k] + b1[k], 0.0f);
    hbuf[z * HIDDEN + k] = (_Float16)hv;

    // ---- spherical harmonics, exactly mirroring the reference recurrences --
    const float inv = 1.0f / fmaxf(nrm, 1e-9f);
    const float x = rx * inv, y = ry * inv, zc = rz * inv;
    const float rxy  = sqrtf(fmaxf(x * x + y * y, 1e-18f));
    const float cphi = x / rxy, sphi = y / rxy;
    const float somx2 = sqrtf(fmaxf(1.0f - zc * zc, 1e-12f));

    float P[5][5];
    float pmm = 1.0f;
    #pragma unroll
    for (int m = 0; m <= 4; ++m) {
        if (m > 0) pmm *= (float)(2 * m - 1) * somx2;
        P[m][m] = pmm;
        if (m + 1 <= 4) P[m + 1][m] = zc * (float)(2 * m + 1) * pmm;
        for (int l = m + 2; l <= 4; ++l)
            P[l][m] = ((float)(2 * l - 1) * zc * P[l - 1][m]
                       - (float)(l + m - 1) * P[l - 2][m]) / (float)(l - m);
    }
    float cm[5], sm[5];
    cm[0] = 1.0f; sm[0] = 0.0f;
    cm[1] = cphi; sm[1] = sphi;
    #pragma unroll
    for (int m = 2; m <= 4; ++m) {
        cm[m] = cm[m - 1] * cphi - sm[m - 1] * sphi;
        sm[m] = sm[m - 1] * cphi + cm[m - 1] * sphi;
    }
    if (k < 25) {
        const int l  = (k < 1) ? 0 : (k < 4) ? 1 : (k < 9) ? 2 : (k < 16) ? 3 : 4;
        const int m  = k - l * l - l;
        const int am = (m < 0) ? -m : m;
        const float fact[9] = {1.f, 1.f, 2.f, 6.f, 24.f, 120.f, 720.f, 5040.f, 40320.f};
        const float N = sqrtf((float)(2 * l + 1) / FOURPI * fact[l - am] / fact[l + am]);
        float v;
        if (m == 0)      v = N * P[l][0];
        else if (m > 0)  v = 1.41421356237309505f * N * P[l][am] * cm[am];
        else             v = 1.41421356237309505f * N * P[l][am] * sm[am];
        Ybuf[z * 25 + k] = v;
    }
}

// ---------------------------------------------------------------------------
// Kernel 2: transpose + cast w2 (64 x NPATH f32) -> w2t (NPATH x 64 f16)
// ---------------------------------------------------------------------------
__global__ void castT_kernel(const float* __restrict__ w2,
                             _Float16* __restrict__ w2t) {
    const int t = blockIdx.x * 256 + threadIdx.x;   // t < NPATH*64
    const int n = t >> 6;
    const int k = t & 63;
    w2t[t] = (_Float16)w2[(size_t)k * NPATH + n];
}

// ---------------------------------------------------------------------------
// Kernel 3: coefficients = h @ w2 + b2 via v_wmma_f32_16x16x32_f16
// One 16x16 C tile per wave; K=64 -> 2 WMMAs with f32 accumulation.
// grid = (NPATH/16/8, BATCH/16), block = 256 (8 waves)
// ---------------------------------------------------------------------------
__global__ void gemm_kernel(const _Float16* __restrict__ hbuf,
                            const _Float16* __restrict__ w2t,
                            const float* __restrict__ b2,
                            float* __restrict__ coeff) {
    const int lane = threadIdx.x & 31;
    const int wave = threadIdx.x >> 5;
    const int m0 = blockIdx.y * 16;
    const int n0 = (blockIdx.x * 8 + wave) * 16;
    const int l16  = lane & 15;
    const int hi   = lane >> 4;          // 0 for lanes 0-15, 1 for lanes 16-31

    // A fragment (16x32, 16-bit): lanes 0-15 hold K=0..15 contiguous,
    // lanes 16-31 hold K=16..31 (matches B)
    const _Float16* ap = hbuf + (m0 + l16) * HIDDEN + hi * 16;
    half8 a0l = *(const half8*)(ap);          // K base+0 .. base+7
    half8 a0h = *(const half8*)(ap + 8);      // K base+8 .. base+15
    half8 a1l = *(const half8*)(ap + 32);
    half8 a1h = *(const half8*)(ap + 40);
    half16 A0 = __builtin_shufflevector(a0l, a0h, 0,1,2,3,4,5,6,7,8,9,10,11,12,13,14,15);
    half16 A1 = __builtin_shufflevector(a1l, a1h, 0,1,2,3,4,5,6,7,8,9,10,11,12,13,14,15);

    // B fragment (32x16, 16-bit): lanes 0-15 hold K=0..15 contiguous,
    // lanes 16-31 hold K=16..31 (same contiguous order as A)
    const _Float16* bp = w2t + (size_t)(n0 + l16) * HIDDEN + hi * 16;
    half16 B0 = *(const half16*)(bp);
    half16 B1 = *(const half16*)(bp + 32);

    float8 acc = {0.f, 0.f, 0.f, 0.f, 0.f, 0.f, 0.f, 0.f};
    acc = __builtin_amdgcn_wmma_f32_16x16x32_f16(false, A0, false, B0,
                                                 (short)0, acc, false, false);
    acc = __builtin_amdgcn_wmma_f32_16x16x32_f16(false, A1, false, B1,
                                                 (short)0, acc, false, false);

    // C/D layout: VGPR g -> row (g + 8*hi), col = lane%16
    const int   p    = n0 + l16;
    const float bias = b2[p];
    float* cp = coeff + (size_t)(m0 + hi * 8) * NPATH + p;
    #pragma unroll
    for (int g = 0; g < 8; ++g)
        cp[(size_t)g * NPATH] = acc[g] + bias;
}

// ---------------------------------------------------------------------------
// Kernel 4: assemble output.  One workgroup per batch element.
// Stage xnorm-scaled QY[io,ji,k] = xn * sum_m Q[io,ji,offq+m]*Y[lf,m] in LDS,
// then each thread owns (u,v) pairs: load nlf coeffs once, fan out to
// (2lo+1)*(2li+1) output elements.
// ---------------------------------------------------------------------------
template<int LO, int LI, int LMIN, int NLF, int QYOFF>
__device__ inline void stage_qy(const float* __restrict__ Q,
                                const float* __restrict__ Yz,
                                float xn, float* __restrict__ qys,
                                int tid, int nthr) {
    constexpr int DO = 2 * LO + 1, DI = 2 * LI + 1;
    constexpr int QW = NLF * (2 * LMIN + NLF);     // total last dim of Q_ij
    constexpr int N  = DO * DI * NLF;
    for (int e = tid; e < N; e += nthr) {
        const int k  = e % NLF;
        const int ji = (e / NLF) % DI;
        const int io = e / (NLF * DI);
        const int lf   = LMIN + k;
        const int offq = k * (2 * LMIN + k);       // sum_{t<k} (2(LMIN+t)+1)
        const int offy = lf * lf;
        const float* Qrow = Q + (io * DI + ji) * QW + offq;
        float s = 0.f;
        for (int m = 0; m < 2 * lf + 1; ++m)
            s += Qrow[m] * Yz[offy + m];
        qys[QYOFF + e] = s * xn;
    }
}

template<int LO, int LI, int MO, int MI, int NLF, int QYOFF, int COFF, int ROW0, int COL0>
__device__ inline void block_out(const float* __restrict__ qys,
                                 const float* __restrict__ cz,
                                 float* __restrict__ outz,
                                 int tid, int nthr) {
    constexpr int DO = 2 * LO + 1, DI = 2 * LI + 1;
    const float* qy = qys + QYOFF;
    const float* cb = cz + COFF;
    for (int e = tid; e < MO * MI; e += nthr) {
        const int u = e / MI;
        const int v = e - u * MI;
        float c[NLF];
        #pragma unroll
        for (int k = 0; k < NLF; ++k) c[k] = cb[e * NLF + k];
        #pragma unroll
        for (int io = 0; io < DO; ++io) {
            float* orow = outz + (size_t)(ROW0 + u * DO + io) * OUTDIM + COL0 + v * DI;
            #pragma unroll
            for (int ji = 0; ji < DI; ++ji) {
                float acc = 0.f;
                #pragma unroll
                for (int k = 0; k < NLF; ++k)
                    acc += qy[(io * DI + ji) * NLF + k] * c[k];
                orow[ji] = acc;
            }
        }
    }
}

__global__ void assemble_kernel(const float* __restrict__ coeff,
                                const float* __restrict__ Ybuf,
                                float* __restrict__ out,
                                const float* __restrict__ Q00, const float* __restrict__ Q01,
                                const float* __restrict__ Q02, const float* __restrict__ Q10,
                                const float* __restrict__ Q11, const float* __restrict__ Q12,
                                const float* __restrict__ Q20, const float* __restrict__ Q21,
                                const float* __restrict__ Q22) {
    const int z   = blockIdx.x;
    const int tid = threadIdx.x;
    __shared__ float Yz[25];
    __shared__ float qys[259];

    if (tid < 25) Yz[tid] = Ybuf[z * 25 + tid];
    __syncthreads();

    // xnorm(i,j) = sqrt((2*li+1) * 4pi / num_summed_i);  num_summed = {224,416,480}
    //            <LO,LI,LMIN,NLF,QYOFF>
    stage_qy<0, 0, 0, 1,   0>(Q00, Yz, sqrtf(1.f * FOURPI / 224.f), qys, tid, 256);
    stage_qy<0, 1, 1, 1,   1>(Q01, Yz, sqrtf(3.f * FOURPI / 224.f), qys, tid, 256);
    stage_qy<0, 2, 2, 1,   4>(Q02, Yz, sqrtf(5.f * FOURPI / 224.f), qys, tid, 256);
    stage_qy<1, 0, 1, 1,   9>(Q10, Yz, sqrtf(1.f * FOURPI / 416.f), qys, tid, 256);
    stage_qy<1, 1, 0, 3,  12>(Q11, Yz, sqrtf(3.f * FOURPI / 416.f), qys, tid, 256);
    stage_qy<1, 2, 1, 3,  39>(Q12, Yz, sqrtf(5.f * FOURPI / 416.f), qys, tid, 256);
    stage_qy<2, 0, 2, 1,  84>(Q20, Yz, sqrtf(1.f * FOURPI / 480.f), qys, tid, 256);
    stage_qy<2, 1, 1, 3,  89>(Q21, Yz, sqrtf(3.f * FOURPI / 480.f), qys, tid, 256);
    stage_qy<2, 2, 0, 5, 134>(Q22, Yz, sqrtf(5.f * FOURPI / 480.f), qys, tid, 256);
    __syncthreads();

    const float* cz   = coeff + (size_t)z * NPATH;
    float*       outz = out   + (size_t)z * OUTDIM * OUTDIM;

    //        <LO,LI, MO, MI,NLF,QYOFF, COFF, ROW0,COL0>
    block_out<0, 0, 128, 128, 1,   0,     0,    0,   0>(qys, cz, outz, tid, 256);
    block_out<0, 1, 128,  64, 1,   1, 16384,    0, 128>(qys, cz, outz, tid, 256);
    block_out<0, 2, 128,  32, 1,   4, 24576,    0, 320>(qys, cz, outz, tid, 256);
    block_out<1, 0,  64, 128, 1,   9, 28672,  128,   0>(qys, cz, outz, tid, 256);
    block_out<1, 1,  64,  64, 3,  12, 36864,  128, 128>(qys, cz, outz, tid, 256);
    block_out<1, 2,  64,  32, 3,  39, 49152,  128, 320>(qys, cz, outz, tid, 256);
    block_out<2, 0,  32, 128, 1,  84, 55296,  320,   0>(qys, cz, outz, tid, 256);
    block_out<2, 1,  32,  64, 3,  89, 59392,  320, 128>(qys, cz, outz, tid, 256);
    block_out<2, 2,  32,  32, 5, 134, 65536,  320, 320>(qys, cz, outz, tid, 256);
}

// ---------------------------------------------------------------------------
// Host-side launch
// ---------------------------------------------------------------------------
extern "C" void kernel_launch(void* const* d_in, const int* in_sizes, int n_in,
                              void* d_out, int out_size, void* d_ws, size_t ws_size,
                              hipStream_t stream) {
    (void)in_sizes; (void)n_in; (void)out_size; (void)ws_size;
    const float* r   = (const float*)d_in[0];
    const float* w1  = (const float*)d_in[1];
    const float* b1  = (const float*)d_in[2];
    const float* w2  = (const float*)d_in[3];
    const float* b2  = (const float*)d_in[4];
    const float* Q00 = (const float*)d_in[5];
    const float* Q01 = (const float*)d_in[6];
    const float* Q02 = (const float*)d_in[7];
    const float* Q10 = (const float*)d_in[8];
    const float* Q11 = (const float*)d_in[9];
    const float* Q12 = (const float*)d_in[10];
    const float* Q20 = (const float*)d_in[11];
    const float* Q21 = (const float*)d_in[12];
    const float* Q22 = (const float*)d_in[13];

    // workspace layout (all offsets 256B-aligned)
    char* ws = (char*)d_ws;
    _Float16* hbuf  = (_Float16*)(ws + 0);                 // 512*64 f16   = 64 KB
    float*    Ybuf  = (float*)   (ws + 65536);             // 512*25 f32   = 50 KB
    _Float16* w2t   = (_Float16*)(ws + 116736);            // NPATH*64 f16 = 8.6 MB
    float*    coeff = (float*)   (ws + 9160704);           // 512*NPATH f32 = 138 MB

    prep_kernel<<<BATCH, HIDDEN, 0, stream>>>(r, w1, b1, hbuf, Ybuf);
    castT_kernel<<<(NPATH * HIDDEN) / 256, 256, 0, stream>>>(w2, w2t);
    gemm_kernel<<<dim3(NPATH / 128, BATCH / 16), 256, 0, stream>>>(hbuf, w2t, b2, coeff);
    assemble_kernel<<<BATCH, 256, 0, stream>>>(coeff, Ybuf, (float*)d_out,
                                               Q00, Q01, Q02, Q10, Q11, Q12,
                                               Q20, Q21, Q22);
}
